// PyramidModuleV1_58669253263760
// MI455X (gfx1250) — matrix operand order repaired and checked by reading, not compiled, not run on hardware
//
#include <hip/hip_runtime.h>
#include <hip/hip_bf16.h>
#include <math.h>

typedef __attribute__((ext_vector_type(16))) _Float16 v16h;
typedef __attribute__((ext_vector_type(8)))  _Float16 v8h;
typedef __attribute__((ext_vector_type(8)))  float    v8f;

#define NPTS_TOT 16384          // B*N_PTS (reference flattens batches)
#define M_TOT    4096           // B*M_ROIS
#define C_IN_    128
#define CG       131
#define CGP      160            // K padded to 5*32 for WMMA
#define O_CH     256
#define NS       32
#define HEADS    4
#define R2f      (1.6f*1.6f)
#define BN_INV   0.99999500003749968f   // 1/sqrt(1+1e-5)

#define PACK1_HALFS (16*5*32*16)   // 40960: key_w / val_w1 (KC=5)
#define PACK2_HALFS (16*8*32*16)   // 65536: val_w2 (KC=8)

#define WMMA_F16(A, B, C) __builtin_amdgcn_wmma_f32_16x16x32_f16(false, (A), false, (B), (short)0, (C), false, false)

// ---------------------------------------------------------------------------
// WMMA B-operand swizzle: element (k, n) of a [K x 32] B matrix lives at
// halfword index ((kc*2 + nt)*32 + lane)*16 + e with
//   kc = k>>5, kk = k&31, nt = n>>4, lane = (n&15)|(kk&16), e = kk&15
// so each lane's 16-half fragment is one contiguous, 32B-aligned LDS run.
// ---------------------------------------------------------------------------
__device__ __forceinline__ int bswz(int k, int n)
{
    int kc = k >> 5, kk = k & 31;
    int lane = (n & 15) | (kk & 16);
    return (((kc * 2 + (n >> 4)) * 32 + lane) << 4) + (kk & 15);
}

// ---------------------------------------------------------------------------
// K1: repack f32 weights into per-lane WMMA A-fragment order (f16).
// A 16x32 f16 layout (contiguous k, matches bswz B): lane<16: M=lane, K=e
//                                  lane>=16: M=lane-16, K=e+16
// ---------------------------------------------------------------------------
__global__ void pack_weights(const float* __restrict__ key_w,
                             const float* __restrict__ val_w1,
                             const float* __restrict__ val_w2,
                             _Float16* __restrict__ pk,
                             _Float16* __restrict__ pv1,
                             _Float16* __restrict__ pv2)
{
    int i = blockIdx.x * blockDim.x + threadIdx.x;
    const int total = 2 * PACK1_HALFS + PACK2_HALFS;
    if (i >= total) return;
    const float* src; _Float16* dst; int KC, W, flat;
    if (i < PACK1_HALFS)            { src = key_w;  dst = pk;  KC = 5; W = CG;   flat = i; }
    else if (i < 2 * PACK1_HALFS)   { src = val_w1; dst = pv1; KC = 5; W = CG;   flat = i - PACK1_HALFS; }
    else                            { src = val_w2; dst = pv2; KC = 8; W = O_CH; flat = i - 2 * PACK1_HALFS; }
    int e    = flat & 15;
    int t    = flat >> 4;
    int lane = t & 31;  t >>= 5;
    int kc   = t % KC;
    int mt   = t / KC;
    int row  = mt * 16 + (lane & 15);
    int k    = (lane < 16) ? e : e + 16;
    k += kc * 32;
    float v = (k < W) ? src[row * W + k] : 0.0f;
    dst[flat] = (_Float16)v;
}

// ---------------------------------------------------------------------------
// K2: ball query — one wave32 per query; append lowest in-radius indices in
// ascending order via ballot + prefix popcount; early exit at 32.
// ---------------------------------------------------------------------------
__global__ __launch_bounds__(256) void ball_query(const float* __restrict__ xyz,
                                                  const float* __restrict__ new_xyz,
                                                  int* __restrict__ idx_out)
{
    __shared__ int lists[8][NS];
    const int tid = threadIdx.x, wave = tid >> 5, lane = tid & 31;
    const int m = blockIdx.x * 8 + wave;
    const float qx = new_xyz[m * 3 + 0];
    const float qy = new_xyz[m * 3 + 1];
    const float qz = new_xyz[m * 3 + 2];
    int count = 0;
    for (int base = 0; base < NPTS_TOT && count < NS; base += 32) {
        int p = base + lane;
        float dx = xyz[p * 3 + 0] - qx;
        float dy = xyz[p * 3 + 1] - qy;
        float dz = xyz[p * 3 + 2] - qz;
        bool in = (dx * dx + dy * dy + dz * dz) < R2f;
        unsigned mask = (unsigned)__ballot(in);
        int pos = count + __popc(mask & ((1u << lane) - 1u));
        if (in && pos < NS) lists[wave][pos] = p;
        count += __popc(mask);
    }
    if (count > NS) count = NS;
    __syncthreads();
    int v = 0;                                  // empty group -> index 0 (matches ref)
    if (count > 0) v = (lane < count) ? lists[wave][lane] : lists[wave][0];
    idx_out[m * NS + lane] = v;
}

// ---------------------------------------------------------------------------
// K3: fully fused per-ROI pipeline. One block (8 waves) per group m.
// Register-blocked WMMA: B fragments loaded once into registers and reused by
// both M-tiles and both GEMMs (key/v); A fragments loaded once per M-tile and
// reused by both N-tiles.
// ---------------------------------------------------------------------------
__global__ __launch_bounds__(256) void fused_attn(
    const float* __restrict__ xyz, const float* __restrict__ new_xyz,
    const float* __restrict__ features, const float* __restrict__ pos_w,
    const float* __restrict__ bn1_g, const float* __restrict__ bn1_b,
    const float* __restrict__ bn2_g, const float* __restrict__ bn2_b,
    const float* __restrict__ attn_w,
    const float* __restrict__ kc_w, const float* __restrict__ qc_w,
    const float* __restrict__ qkc_w, const float* __restrict__ vc_w,
    const float* __restrict__ cls_w1, const float* __restrict__ cls_b1,
    const float* __restrict__ cls_w2, const float* __restrict__ cls_b2,
    const float* __restrict__ reg_w1, const float* __restrict__ reg_b1,
    const float* __restrict__ reg_w2, const float* __restrict__ reg_b2,
    const int* __restrict__ idx_in,
    const _Float16* __restrict__ packKey, const _Float16* __restrict__ packV1,
    const _Float16* __restrict__ packV2,
    float* __restrict__ out_nf, float* __restrict__ out_cls, float* __restrict__ out_reg)
{
    __shared__ __align__(32) _Float16 gfB[CGP * NS];   // B-fragment order (5 kc x 2 nt)
    __shared__ __align__(32) _Float16 vB [O_CH * NS];  // B-fragment order (8 kc x 2 nt)
    __shared__ float gxS [3 * NS];
    __shared__ float posS[O_CH * NS];
    __shared__ float keyS[O_CH * NS];
    __shared__ float valS[O_CH * NS];
    __shared__ float meanPos[O_CH], meanKey[O_CH], meanPK[O_CH];
    __shared__ float gVC[O_CH], gQC[O_CH], gKC[O_CH], gQKC[O_CH];
    __shared__ float attnS[HEADS * NS];
    __shared__ float nfS[O_CH];
    __shared__ float h1c[128], h1r[128];
    __shared__ float gateS[2];
    __shared__ int   idxS[NS];

    const int m    = blockIdx.x;
    const int tid  = threadIdx.x;
    const int wave = tid >> 5, lane = tid & 31;

    if (tid < NS) idxS[tid] = idx_in[m * NS + tid];
    __syncthreads();

    // ---- gather: gx rows 0..2, features rows 3..130, zero pad to 160 ----
    const float q0 = new_xyz[m * 3 + 0], q1 = new_xyz[m * 3 + 1], q2 = new_xyz[m * 3 + 2];
    if (tid < NS) {
        int p = idxS[tid];
        float g0 = xyz[p * 3 + 0] - q0, g1 = xyz[p * 3 + 1] - q1, g2 = xyz[p * 3 + 2] - q2;
        gxS[0 * NS + tid] = g0; gxS[1 * NS + tid] = g1; gxS[2 * NS + tid] = g2;
        gfB[bswz(0, tid)] = (_Float16)g0;
        gfB[bswz(1, tid)] = (_Float16)g1;
        gfB[bswz(2, tid)] = (_Float16)g2;
    }
    for (int i = tid; i < (CGP - 3) * NS; i += 256) {
        int k = 3 + (i >> 5), n = i & 31;
        float v = 0.f;
        if (k < CG) {
            int p = idxS[n];
            int b = p >> 13, pp = p & 8191;          // pidx -> (batch, point)
            v = features[(b * C_IN_ + (k - 3)) * 8192 + pp];
        }
        gfB[bswz(k, n)] = (_Float16)v;
    }
    __syncthreads();

    // ---- pos = relu(pos_w @ gx): K=3, pure VALU ----
    for (int i = tid; i < O_CH * NS; i += 256) {
        int o = i >> 5, n = i & 31;
        float s = pos_w[o * 3 + 0] * gxS[n] + pos_w[o * 3 + 1] * gxS[NS + n]
                + pos_w[o * 3 + 2] * gxS[2 * NS + n];
        posS[i] = fmaxf(s, 0.f);
    }

    const int n0 = (lane & 15), n1 = 16 + (lane & 15);

    // ---- GEMM1+2: key = relu(key_w @ gf), v = relu(bn1(val_w1 @ gf)) ----
    {
        v16h bf[5][2];                 // all B fragments, loaded once (80 VGPRs)
#pragma unroll
        for (int kc = 0; kc < 5; ++kc)
#pragma unroll
            for (int nt = 0; nt < 2; ++nt)
                bf[kc][nt] = *(const v16h*)(gfB + (((kc * 2 + nt) * 32 + lane) << 4));

        for (int mtl = 0; mtl < 2; ++mtl) {
            int mt = wave * 2 + mtl;
            v8f accK0 = {}, accK1 = {}, accV0 = {}, accV1 = {};
#pragma unroll
            for (int kc = 0; kc < 5; ++kc) {
                v16h aK = *(const v16h*)(packKey + ((mt * 5 + kc) * 32 + lane) * 16);
                v16h aV = *(const v16h*)(packV1  + ((mt * 5 + kc) * 32 + lane) * 16);
                accK0 = WMMA_F16(aK, bf[kc][0], accK0);
                accK1 = WMMA_F16(aK, bf[kc][1], accK1);
                accV0 = WMMA_F16(aV, bf[kc][0], accV0);
                accV1 = WMMA_F16(aV, bf[kc][1], accV1);
            }
            int mr = mt * 16 + ((lane & 16) ? 8 : 0);
#pragma unroll
            for (int r = 0; r < 8; ++r) {
                keyS[(mr + r) * NS + n0] = fmaxf(accK0[r], 0.f);
                keyS[(mr + r) * NS + n1] = fmaxf(accK1[r], 0.f);
            }
            v8h p0, p1;
#pragma unroll
            for (int r = 0; r < 8; ++r) {
                int M = mr + r;
                float g = bn1_g[M], bb = bn1_b[M];
                p0[r] = (_Float16)fmaxf(g * (accV0[r] * BN_INV) + bb, 0.f);
                p1[r] = (_Float16)fmaxf(g * (accV1[r] * BN_INV) + bb, 0.f);
            }
            // rows mr..mr+7 map to 8 consecutive halfword slots -> one 16B store
            *(v8h*)(vB + bswz(mr, n0)) = p0;
            *(v8h*)(vB + bswz(mr, n1)) = p1;
        }
    }
    __syncthreads();   // all of vB needed (K=256 spans every wave's M tiles)

    // ---- GEMM3: val = relu(bn2(val_w2 @ v))  (WMMA f16, K=256) ----
    {
        v16h bf[8][2];                 // all B fragments, loaded once (128 VGPRs)
#pragma unroll
        for (int kc = 0; kc < 8; ++kc)
#pragma unroll
            for (int nt = 0; nt < 2; ++nt)
                bf[kc][nt] = *(const v16h*)(vB + (((kc * 2 + nt) * 32 + lane) << 4));

        for (int mtl = 0; mtl < 2; ++mtl) {
            int mt = wave * 2 + mtl;
            v8f acc0 = {}, acc1 = {};
#pragma unroll
            for (int kc = 0; kc < 8; ++kc) {
                v16h a = *(const v16h*)(packV2 + ((mt * 8 + kc) * 32 + lane) * 16);
                acc0 = WMMA_F16(a, bf[kc][0], acc0);
                acc1 = WMMA_F16(a, bf[kc][1], acc1);
            }
            int mr = mt * 16 + ((lane & 16) ? 8 : 0);
#pragma unroll
            for (int r = 0; r < 8; ++r) {
                int M = mr + r;
                float g = bn2_g[M], bb = bn2_b[M];
                valS[M * NS + n0] = fmaxf(g * (acc0[r] * BN_INV) + bb, 0.f);
                valS[M * NS + n1] = fmaxf(g * (acc1[r] * BN_INV) + bb, 0.f);
            }
        }
    }
    __syncthreads();

    // ---- per-channel means over NS ----
    {
        float sp = 0.f, sk = 0.f, spk = 0.f;
#pragma unroll 4
        for (int n = 0; n < NS; ++n) {
            float p_ = posS[tid * NS + n], k_ = keyS[tid * NS + n];
            sp += p_; sk += k_; spk += p_ * k_;
        }
        meanPos[tid] = sp * (1.f / NS);
        meanKey[tid] = sk * (1.f / NS);
        meanPK[tid]  = spk * (1.f / NS);
    }
    __syncthreads();

    // ---- four sigmoid channel gates (256x256 matvecs) ----
    {
        float avc = 0.f, aqc = 0.f, akc = 0.f, aqkc = 0.f;
        for (int c = 0; c < O_CH; ++c) {
            float mp = meanPos[c], mk = meanKey[c], mpk = meanPK[c];
            avc  += vc_w [tid * O_CH + c] * mp;
            aqc  += qc_w [tid * O_CH + c] * mp;
            akc  += kc_w [tid * O_CH + c] * mk;
            aqkc += qkc_w[tid * O_CH + c] * mpk;
        }
        gVC [tid] = 1.f / (1.f + expf(-avc));
        gQC [tid] = 1.f / (1.f + expf(-aqc));
        gKC [tid] = 1.f / (1.f + expf(-akc));
        gQKC[tid] = 1.f / (1.f + expf(-aqkc));
    }
    __syncthreads();

    // ---- val += pos * v_c ----
    for (int i = tid; i < O_CH * NS; i += 256) {
        int o = i >> 5;
        valS[i] += posS[i] * gVC[o];
    }
    __syncthreads();

    // ---- attention logits (attn_emb recomputed on the fly) + softmax ----
    if (tid < HEADS * NS) {
        int h = tid >> 5, n = tid & 31;
        float acc = 0.f;
        for (int o = 0; o < O_CH; ++o) {
            float p_ = posS[o * NS + n], k_ = keyS[o * NS + n];
            float e = p_ * gQC[o] + k_ * gKC[o] + p_ * k_ * gQKC[o];
            acc += attn_w[h * O_CH + o] * e;
        }
        attnS[h * NS + n] = acc;
    }
    __syncthreads();
    if (tid < HEADS) {
        float mx = -1e30f;
        for (int n = 0; n < NS; ++n) mx = fmaxf(mx, attnS[tid * NS + n]);
        float s = 0.f;
        for (int n = 0; n < NS; ++n) {
            float ex = expf(attnS[tid * NS + n] - mx);
            attnS[tid * NS + n] = ex; s += ex;
        }
        float inv = 1.f / s;
        for (int n = 0; n < NS; ++n) attnS[tid * NS + n] *= inv;
    }
    __syncthreads();

    // ---- new_features: per-head weighted reduce over NS ----
    {
        int h = tid >> 6;                         // o = h*64 + d
        float nf = 0.f;
#pragma unroll 4
        for (int n = 0; n < NS; ++n) nf += valS[tid * NS + n] * attnS[h * NS + n];
        nfS[tid] = nf;
        out_nf[m * O_CH + tid] = nf;
    }
    __syncthreads();

    // ---- cls/reg gating heads (256->128->1, sigmoid) ----
    if (tid < 128) {
        float s = cls_b1[tid];
        for (int o = 0; o < O_CH; ++o) s += cls_w1[tid * O_CH + o] * nfS[o];
        h1c[tid] = fmaxf(s, 0.f);
    } else {
        int j = tid - 128;
        float s = reg_b1[j];
        for (int o = 0; o < O_CH; ++o) s += reg_w1[j * O_CH + o] * nfS[o];
        h1r[j] = fmaxf(s, 0.f);
    }
    __syncthreads();
    if (tid == 0) {
        float s = cls_b2[0];
        for (int j = 0; j < 128; ++j) s += cls_w2[j] * h1c[j];
        gateS[0] = 1.f / (1.f + expf(-s));
    } else if (tid == 32) {
        float s = reg_b2[0];
        for (int j = 0; j < 128; ++j) s += reg_w2[j] * h1r[j];
        gateS[1] = 1.f / (1.f + expf(-s));
    }
    __syncthreads();
    out_cls[m * O_CH + tid] = nfS[tid] * gateS[0];
    out_reg[m * O_CH + tid] = nfS[tid] * gateS[1];
}

// ---------------------------------------------------------------------------
extern "C" void kernel_launch(void* const* d_in, const int* in_sizes, int n_in,
                              void* d_out, int out_size, void* d_ws, size_t ws_size,
                              hipStream_t stream) {
    (void)in_sizes; (void)n_in; (void)out_size; (void)ws_size;
    const float* xyz     = (const float*)d_in[0];
    const float* new_xyz = (const float*)d_in[1];
    const float* feats   = (const float*)d_in[2];
    const float* pos_w   = (const float*)d_in[3];
    const float* key_w   = (const float*)d_in[4];
    const float* val_w1  = (const float*)d_in[5];
    const float* bn1_g   = (const float*)d_in[6];
    const float* bn1_b   = (const float*)d_in[7];
    const float* val_w2  = (const float*)d_in[8];
    const float* bn2_g   = (const float*)d_in[9];
    const float* bn2_b   = (const float*)d_in[10];
    const float* attn_w  = (const float*)d_in[11];
    const float* kc_w    = (const float*)d_in[12];
    const float* qc_w    = (const float*)d_in[13];
    const float* qkc_w   = (const float*)d_in[14];
    const float* vc_w    = (const float*)d_in[15];
    const float* cls_w1  = (const float*)d_in[16];
    const float* cls_b1  = (const float*)d_in[17];
    const float* cls_w2  = (const float*)d_in[18];
    const float* cls_b2  = (const float*)d_in[19];
    const float* reg_w1  = (const float*)d_in[20];
    const float* reg_b1  = (const float*)d_in[21];
    const float* reg_w2  = (const float*)d_in[22];
    const float* reg_b2  = (const float*)d_in[23];

    // workspace layout: idx[4096*32] | packKey | packV1 | packV2  (~820 KB)
    int*      ws_idx = (int*)d_ws;
    _Float16* pk  = (_Float16*)((char*)d_ws + (size_t)M_TOT * NS * sizeof(int));
    _Float16* pv1 = pk  + PACK1_HALFS;
    _Float16* pv2 = pv1 + PACK1_HALFS;

    float* out     = (float*)d_out;
    float* out_nf  = out;
    float* out_cls = out + (size_t)M_TOT * O_CH;
    float* out_reg = out + 2 * (size_t)M_TOT * O_CH;

    const int pack_total = 2 * PACK1_HALFS + PACK2_HALFS;
    pack_weights<<<(pack_total + 255) / 256, 256, 0, stream>>>(key_w, val_w1, val_w2, pk, pv1, pv2);
    ball_query<<<M_TOT / 8, 256, 0, stream>>>(xyz, new_xyz, ws_idx);
    fused_attn<<<M_TOT, 256, 0, stream>>>(
        xyz, new_xyz, feats, pos_w, bn1_g, bn1_b, bn2_g, bn2_b, attn_w,
        kc_w, qc_w, qkc_w, vc_w, cls_w1, cls_b1, cls_w2, cls_b2,
        reg_w1, reg_b1, reg_w2, reg_b2,
        ws_idx, pk, pv1, pv2, out_nf, out_cls, out_reg);
}
